// SpatialRefinementModule_40802189312220
// MI455X (gfx1250) — compile-verified, matrix operand order checked by reading, not run on hardware
//
#include <hip/hip_runtime.h>

typedef __attribute__((ext_vector_type(16))) _Float16 v16h;
typedef __attribute__((ext_vector_type(4)))  _Float16 v4h;
typedef __attribute__((ext_vector_type(8)))  float    v8f;

#define B_   4
#define N_   4096
#define C_   256
#define M_   2048
#define NT   64          // N-rows per block
#define MT   32          // M-chunk per iteration
#define NWAVES 4         // 128 threads, wave32
#define NTHREADS (NWAVES*32)
#define TAU_ 0.4f
#define EPS_ 1e-8f

union FragH {
    v16h v;
    _Float16 h[16];
    unsigned int d[8];
};

// Load a 16x16x32-f16 WMMA A/B fragment from a row-major f16 LDS matrix.
// A-layout (M x K) and B-layout (K x N, N-major storage) share this indexing:
//   lane L: row = row0 + (L&15); halves 0..7  = K (k0+sel)..(k0+sel+7)
//                                halves 8..15 = K (k0+16+sel)..(k0+16+sel+7)
//   sel = (L<16) ? 0 : 8   ->  two ds_load_b128 per lane.
__device__ __forceinline__ v16h load_frag16(const _Float16* base, int ld,
                                            int row0, int k0, int lane) {
    const _Float16* p = base + (size_t)(row0 + (lane & 15)) * ld + k0 + ((lane & 16) ? 8 : 0);
    FragH f;
    *(uint4*)(&f.d[0]) = *(const uint4*)(p);
    *(uint4*)(&f.d[4]) = *(const uint4*)(p + 16);
    return f.v;
}

__global__ __launch_bounds__(NTHREADS)
void srm_fused_kernel(const float* __restrict__ L,   // [B, N, C]
                      const float* __restrict__ G,   // [B, C, M]
                      float* __restrict__ O) {       // [B, N, C]
    __shared__ __align__(16) float    rawG[C_][MT];          // async-staged f32 g chunk (32KB)
    __shared__ __align__(16) _Float16 G1[MT][C_];            // gT chunk  [m][c]  (GEMM1 B frags)
    __shared__ __align__(16) _Float16 G2[C_][MT];            // g  chunk  [c][m]  (GEMM2 B frags)
    __shared__ __align__(16) _Float16 simBuf[NWAVES][16][MT];// per-wave sim tile (A frags GEMM2)
    __shared__ float gN[M_];      // g column norms for this batch
    __shared__ float lN[NT];      // l row norms for this block
    __shared__ float lpart[NTHREADS];

    const int tid  = threadIdx.x;
    const int lane = tid & 31;
    const int wid  = tid >> 5;
    const int b    = blockIdx.y;
    const int n0   = blockIdx.x * NT;

    // exp(cos/tau) == exp2(cos * log2(e)/tau)
    const float kScale = 1.4426950408889634f / TAU_;

    const float* Lb = L + (size_t)b * N_ * C_;
    const float* Gb = G + (size_t)b * C_ * M_;

    // ---------- g column norms (coalesced: threads sweep m, loop c) ----------
    {
        float s[M_ / NTHREADS];                 // 16 partials per thread
        #pragma unroll
        for (int j = 0; j < M_ / NTHREADS; ++j) s[j] = 0.f;
        for (int c = 0; c < C_; ++c) {
            const float* row = Gb + (size_t)c * M_;
            #pragma unroll
            for (int j = 0; j < M_ / NTHREADS; ++j) {
                float v = row[tid + j * NTHREADS];
                s[j] += v * v;
            }
        }
        #pragma unroll
        for (int j = 0; j < M_ / NTHREADS; ++j)
            gN[tid + j * NTHREADS] = sqrtf(s[j]);
    }

    // ---------- l row norms for this block's 64 rows ----------
    {
        const int row = tid >> 1;
        const float* lr = Lb + (size_t)(n0 + row) * C_ + (tid & 1) * (C_ / 2);
        float s = 0.f;
        #pragma unroll 8
        for (int i = 0; i < C_ / 2; ++i) { float v = lr[i]; s += v * v; }
        lpart[tid] = s;
    }
    __syncthreads();
    if (tid < NT) lN[tid] = sqrtf(lpart[2 * tid] + lpart[2 * tid + 1]);
    __syncthreads();

    // ---------- resident f16 A-fragments of l for this wave's 16 rows ----------
    const int nw = n0 + wid * 16;
    const float* lrow = Lb + (size_t)(nw + (lane & 15)) * C_;
    const int ksel = (lane & 16) ? 8 : 0;
    v16h lA[C_ / 32];
    #pragma unroll
    for (int kc = 0; kc < C_ / 32; ++kc) {
        FragH a;
        #pragma unroll
        for (int i = 0; i < 8; ++i) a.h[i]     = (_Float16)lrow[kc * 32 + ksel + i];
        #pragma unroll
        for (int i = 0; i < 8; ++i) a.h[8 + i] = (_Float16)lrow[kc * 32 + 16 + ksel + i];
        lA[kc] = a.v;
    }

    // l-norms pre-gathered in C/D-tile layout (VGPR r -> row r / r+8)
    float lnv[8];
    #pragma unroll
    for (int r = 0; r < 8; ++r)
        lnv[r] = lN[wid * 16 + ((lane & 16) ? r + 8 : r)];

    v8f acc[C_ / 16];            // 16 x 256 f32 output accumulator (128 VGPRs)
    v8f vzero = {};
    #pragma unroll
    for (int t = 0; t < C_ / 16; ++t) acc[t] = vzero;
    float rs[8];
    #pragma unroll
    for (int r = 0; r < 8; ++r) rs[r] = 0.f;

    // async copy of one g chunk (32KB) into rawG; wave w covers c-rows
    // [w*64, w*64+64), one b128 per lane, 4 c-rows per instruction.
    // Rolled loop: fire-and-forget issue, keep address math incremental
    // (minimizes live VGPRs -> stay under the 256-VGPR direct window).
    const int cSub = lane >> 3;              // 0..3
    const int mOff = (lane & 7) * 4;         // float index within chunk row
    auto issue_async_chunk = [&](int bm0) {
        const int c0 = wid * (C_ / NWAVES) + cSub;
        const float* ga = Gb + (size_t)c0 * M_ + bm0 + mOff;
        unsigned lo = (unsigned)(size_t)&rawG[c0][mOff];
        #pragma unroll 1
        for (int i = 0; i < (C_ / NWAVES) / 4; ++i) {      // 16 instrs/wave
            asm volatile("global_load_async_to_lds_b128 %0, %1, off"
                         :: "v"(lo), "v"(ga) : "memory");
            ga += (size_t)4 * M_;            // +4 c-rows in global
            lo += 4 * MT * 4;                // +4 c-rows in rawG
        }
    };

    // prologue: prefetch chunk 0
    issue_async_chunk(0);

    // =============== main fused loop over M in chunks of MT ===============
    for (int mc = 0; mc < M_ / MT; ++mc) {
        const int bm0 = mc * MT;
        asm volatile("s_wait_asynccnt 0x0" ::: "memory");  // rawG chunk mc landed
        __syncthreads();   // all waves' async done + prev chunk's G1/G2 reads done

        // convert rawG -> G1[m][c] and G2[c][m] (f32 -> f16)
        // (unroll capped to bound live LDS-load/cvt temps)
        #pragma unroll 2
        for (int idx = tid; idx < C_ * MT / 4; idx += NTHREADS) {
            const int c = idx >> 3;               // MT/4 = 8 float4 per c-row
            const int m = (idx & 7) * 4;
            float4 v = *(const float4*)(&rawG[c][m]);
            _Float16 h0 = (_Float16)v.x, h1 = (_Float16)v.y;
            _Float16 h2 = (_Float16)v.z, h3 = (_Float16)v.w;
            v4h pk = {h0, h1, h2, h3};
            *(v4h*)(&G2[c][m]) = pk;
            G1[m + 0][c] = h0; G1[m + 1][c] = h1;
            G1[m + 2][c] = h2; G1[m + 3][c] = h3;
        }
        __syncthreads();   // rawG drained, G1/G2 ready

        // prefetch next chunk; DMA overlaps the whole GEMM1+GEMM2 body below
        if (mc + 1 < M_ / MT) issue_async_chunk(bm0 + MT);

        // ---- GEMM1: dot[16 x MT] = l_tile x g_chunk, K = C = 256 ----
        #pragma unroll
        for (int mt = 0; mt < MT / 16; ++mt) {
            v8f d = vzero;
            #pragma unroll
            for (int kc = 0; kc < C_ / 32; ++kc) {
                v16h bf = load_frag16(&G1[0][0], C_, mt * 16, kc * 32, lane);
                d = __builtin_amdgcn_wmma_f32_16x16x32_f16(
                        false, lA[kc], false, bf, (short)0, d, false, false);
            }
            // ---- epilogue: cosine -> exp2 -> rowsum, stash sim as f16 ----
            const float gn = gN[bm0 + mt * 16 + (lane & 15)];
            #pragma unroll
            for (int r = 0; r < 8; ++r) {
                float denom = fmaxf(lnv[r] * gn, EPS_);
                float inv   = __builtin_amdgcn_rcpf(denom);
                float simv  = __builtin_amdgcn_exp2f(d[r] * inv * kScale);
                _Float16 sh = (_Float16)simv;
                rs[r] += (float)sh;   // keep rowsum consistent with f16 A of GEMM2
                simBuf[wid][(lane & 16) ? r + 8 : r][mt * 16 + (lane & 15)] = sh;
            }
        }

        // ---- GEMM2: acc[16 x 256] += sim[16 x MT] x gT[MT x 256] ----
        // (D-layout -> A-layout transpose via per-wave LDS staging;
        //  wave-internal DScnt ordering, no barrier needed)
        v16h sA[MT / 32];
        #pragma unroll
        for (int kk = 0; kk < MT / 32; ++kk)
            sA[kk] = load_frag16(&simBuf[wid][0][0], MT, 0, kk * 32, lane);
        #pragma unroll
        for (int ct = 0; ct < C_ / 16; ++ct) {
            #pragma unroll
            for (int kk = 0; kk < MT / 32; ++kk) {
                v16h bf = load_frag16(&G2[0][0], MT, ct * 16, kk * 32, lane);
                acc[ct] = __builtin_amdgcn_wmma_f32_16x16x32_f16(
                              false, sA[kk], false, bf, (short)0, acc[ct], false, false);
            }
        }
    }

    // ---------- rowsum: butterfly reduce across each 16-lane half ----------
    float rsinv[8];
    #pragma unroll
    for (int r = 0; r < 8; ++r) {
        float v = rs[r];
        v += __shfl_xor(v, 1, 32);
        v += __shfl_xor(v, 2, 32);
        v += __shfl_xor(v, 4, 32);
        v += __shfl_xor(v, 8, 32);
        rsinv[r] = __builtin_amdgcn_rcpf(v);
    }

    // ---------- out = l + acc * (1/rowsum) ----------
    #pragma unroll
    for (int ct = 0; ct < C_ / 16; ++ct) {
        #pragma unroll
        for (int r = 0; r < 8; ++r) {
            const int row = (lane & 16) ? r + 8 : r;
            const int n   = nw + row;
            const int c   = ct * 16 + (lane & 15);
            const size_t off = ((size_t)b * N_ + n) * C_ + c;
            O[off] = L[off] + acc[ct][r] * rsinv[r];
        }
    }
}

extern "C" void kernel_launch(void* const* d_in, const int* in_sizes, int n_in,
                              void* d_out, int out_size, void* d_ws, size_t ws_size,
                              hipStream_t stream) {
    (void)in_sizes; (void)n_in; (void)out_size; (void)d_ws; (void)ws_size;
    const float* l_feat = (const float*)d_in[0];   // [4, 4096, 256]
    const float* g_feat = (const float*)d_in[1];   // [4, 256, 2048]
    float* out = (float*)d_out;                    // [4, 4096, 256]

    dim3 grid(N_ / NT, B_);     // 64 x 4 = 256 workgroups
    dim3 block(NTHREADS);       // 128 threads = 4 waves (wave32)
    srm_fused_kernel<<<grid, block, 0, stream>>>(l_feat, g_feat, out);
}
